// MultiTaskModel_46926812676625
// MI455X (gfx1250) — matrix operand order repaired and checked by reading, not compiled, not run on hardware
//
#include <hip/hip_runtime.h>
#include <math.h>

// ---------------------------------------------------------------------------
// Types for CDNA5 WMMA (gfx1250, wave32)
// ---------------------------------------------------------------------------
typedef __attribute__((ext_vector_type(16))) __bf16 v16bf;
typedef __attribute__((ext_vector_type(8)))  float  v8f;

__device__ inline unsigned short f2bfbits(float f) {
    // round-to-nearest-even f32 -> bf16 bit pattern
    unsigned u = __float_as_uint(f);
    return (unsigned short)((u + 0x7FFFu + ((u >> 16) & 1u)) >> 16);
}

__device__ inline float wave_reduce_sum(float v) {
#pragma unroll
    for (int m = 16; m >= 1; m >>= 1) v += __shfl_xor(v, m, 32);
    return v;
}

__device__ inline void atomicMaxF(float* addr, float val) {
    int* ai = (int*)addr;
    int old = *ai;
    while (true) {
        float of = __int_as_float(old);
        if (of >= val) break;
        int assumed = old;
        old = atomicCAS(ai, assumed, __float_as_int(val));
        if (old == assumed) break;
    }
}

// ---------------------------------------------------------------------------
// Precision-conversion kernels (run once per GEMM; weights fit in 192MB L2)
// ---------------------------------------------------------------------------
// A [M,K] f32 row-major -> [M,Ka] bf16 row-major, zero-padded K..Ka
__global__ void conv_a(const float* __restrict__ A, unsigned short* __restrict__ O,
                       int K, int Ka, size_t total) {
    size_t i = (size_t)blockIdx.x * blockDim.x + threadIdx.x;
    if (i >= total) return;
    int kk = (int)(i % Ka);
    size_t m = i / Ka;
    O[i] = (kk < K) ? f2bfbits(A[m * K + kk]) : (unsigned short)0;
}

// W [K,Nc] f32 -> Wt [Nc,Ka] bf16 (transposed), zero-padded K..Ka. Tiled via LDS.
__global__ void conv_wt(const float* __restrict__ W, unsigned short* __restrict__ Wt,
                        int K, int Nc, int Ka) {
    __shared__ float t[32][33];
    int n0 = blockIdx.x * 32, k0 = blockIdx.y * 32;
    int tx = threadIdx.x, ty = threadIdx.y;   // blockDim = (32, 8)
#pragma unroll
    for (int r = 0; r < 32; r += 8) {
        int k = k0 + ty + r, n = n0 + tx;
        t[ty + r][tx] = (k < K && n < Nc) ? W[(size_t)k * Nc + n] : 0.f;
    }
    __syncthreads();
#pragma unroll
    for (int r = 0; r < 32; r += 8) {
        int n = n0 + ty + r, k = k0 + tx;
        if (n < Nc && k < Ka) Wt[(size_t)n * Ka + k] = f2bfbits(t[tx][ty + r]);
    }
}

// ---------------------------------------------------------------------------
// Guard-free tiled GEMM: C[M,Nc] = act(A @ W + bias).
// A16: [M,Ka] bf16 row-major. Wt16: [Nc,Ka] bf16 (W transposed).
// Requires M%64==0, Nc%64==0, Ka%32==0 (guaranteed by the launcher).
// Block = 128 threads (4 waves), tile 64x64, wave tile 32x32 = 2x2 WMMA.
// LDS holds tiles in fragment-swizzled order: [subtile][lane][32B], so each
// lane's WMMA operand is one contiguous 32B LDS read (2 x ds_load_b128).
// Register double-buffering hides global latency behind the WMMAs.
// ---------------------------------------------------------------------------
__global__ __launch_bounds__(128) void gemm_bf16_wmma(
    const unsigned short* __restrict__ A16, const unsigned short* __restrict__ Wt16,
    float* __restrict__ C, const float* __restrict__ bias,
    int Nc, int Ka, int ldc, int act)
{
    __shared__ uint4 As[4][32][2];   // 4KB: A tile, fragment order
    __shared__ uint4 Bs[4][32][2];   // 4KB: B tile, fragment order

    const int tid  = threadIdx.x;
    const int lane = tid & 31;
    const int wave = tid >> 5;
    const int bm = blockIdx.y * 64;
    const int bn = blockIdx.x * 64;
    const int wr = (wave >> 1) * 32;
    const int wc = (wave & 1) * 32;
    const int ia0 = wr >> 4;         // first A subtile for this wave
    const int jb0 = wc >> 4;         // first B subtile for this wave

    // staging roles: thread t handles row rr (A) / column rr (Wt), k-half sh
    const int rr = tid >> 1, sh = tid & 1;
    const int i_sub = rr >> 4, r_sub = rr & 15;
    const int b_lane = sh * 16 + (rr & 15), j_sub = rr >> 4;

    const uint4* Arow = (const uint4*)(A16  + (size_t)(bm + rr) * Ka);
    const uint4* Brow = (const uint4*)(Wt16 + (size_t)(bn + rr) * Ka);

    v8f acc[2][2];
#pragma unroll
    for (int i = 0; i < 2; i++)
#pragma unroll
        for (int j = 0; j < 2; j++)
#pragma unroll
            for (int e = 0; e < 8; e++) acc[i][j][e] = 0.f;

    // prefetch first k-tile into registers (32B A + 32B B per thread)
    int u = 2 * sh;
    uint4 a0 = Arow[u], a1 = Arow[u + 1];
    uint4 b0 = Brow[u], b1 = Brow[u + 1];

    for (int k0 = 0; k0 < Ka; k0 += 32) {
        // commit staged registers to LDS in fragment order
        As[i_sub][r_sub][sh]      = a0;   // k[k0+16sh .. +8)  -> half0 lane
        As[i_sub][16 + r_sub][sh] = a1;   // k[k0+16sh+8 ..+16)-> half1 lane
        Bs[j_sub][r_sub][sh]      = b0;   // k[k0+16sh .. +8)  -> half0 lane
        Bs[j_sub][16 + r_sub][sh] = b1;   // k[k0+16sh+8 ..+16)-> half1 lane
        __syncthreads();

        // prefetch next k-tile while WMMAs run (uniform branch)
        if (k0 + 32 < Ka) {
            int u2 = ((k0 + 32) >> 3) + 2 * sh;
            a0 = Arow[u2]; a1 = Arow[u2 + 1];
            b0 = Brow[u2]; b1 = Brow[u2 + 1];
        }

        // operand fragments: one contiguous 32B read per fragment per lane
        v16bf af0 = *(const v16bf*)&As[ia0][lane][0];
        v16bf af1 = *(const v16bf*)&As[ia0 + 1][lane][0];
        v16bf bf0 = *(const v16bf*)&Bs[jb0][lane][0];
        v16bf bf1 = *(const v16bf*)&Bs[jb0 + 1][lane][0];

        acc[0][0] = __builtin_amdgcn_wmma_f32_16x16x32_bf16(false, af0, false, bf0, (short)0, acc[0][0], false, false);
        acc[0][1] = __builtin_amdgcn_wmma_f32_16x16x32_bf16(false, af0, false, bf1, (short)0, acc[0][1], false, false);
        acc[1][0] = __builtin_amdgcn_wmma_f32_16x16x32_bf16(false, af1, false, bf0, (short)0, acc[1][0], false, false);
        acc[1][1] = __builtin_amdgcn_wmma_f32_16x16x32_bf16(false, af1, false, bf1, (short)0, acc[1][1], false, false);
        __syncthreads();
    }

    // store: C/D layout VGPR v -> M = v + 8*half, N = lane&15 (no guards needed)
    const int half = lane >> 4, l16 = lane & 15;
#pragma unroll
    for (int i = 0; i < 2; i++) {
#pragma unroll
        for (int j = 0; j < 2; j++) {
            int col = bn + wc + j * 16 + l16;
            float bv = bias ? bias[col] : 0.f;
#pragma unroll
            for (int v = 0; v < 8; v++) {
                int row = bm + wr + i * 16 + half * 8 + v;
                float o = acc[i][j][v] + bv;
                if (act == 1) o = o > 0.f ? o : 0.f;
                C[(size_t)row * ldc + col] = o;
            }
        }
    }
}

// ---------------------------------------------------------------------------
// Utility / edge kernels
// ---------------------------------------------------------------------------
__global__ void fillf(float* p, float v, size_t n) {
    size_t i = (size_t)blockIdx.x * blockDim.x + threadIdx.x;
    if (i < n) p[i] = v;
}

__global__ void build_edges(const int* __restrict__ ei, int* src, int* dst, int E, int N) {
    int i = blockIdx.x * blockDim.x + threadIdx.x;
    if (i >= E + N) return;
    if (i < E) { src[i] = ei[i]; dst[i] = ei[E + i]; }
    else       { src[i] = i - E; dst[i] = i - E; }
}

// al_s[n,h] = sum_c H[n,h,c]*a_src[h,c]; al_d likewise. One wave per (n,h), H=8.
__global__ void gat_head_proj(const float* __restrict__ H, const float* __restrict__ as_,
                              const float* __restrict__ ad_, float* als, float* ald,
                              int N, int Co) {
    int gt = blockIdx.x * blockDim.x + threadIdx.x;
    int w = gt >> 5, lane = gt & 31;
    if (w >= N * 8) return;
    int n = w >> 3, h = w & 7;
    const float* hp = H   + (size_t)n * 8 * Co + (size_t)h * Co;
    const float* ap = as_ + (size_t)h * Co;
    const float* dp = ad_ + (size_t)h * Co;
    float ps = 0.f, pd = 0.f;
    for (int c = lane; c < Co; c += 32) { float v = hp[c]; ps += v * ap[c]; pd += v * dp[c]; }
    ps = wave_reduce_sum(ps); pd = wave_reduce_sum(pd);
    if (lane == 0) { als[w] = ps; ald[w] = pd; }
}

__global__ void gat_logits(const int* __restrict__ src, const int* __restrict__ dst,
                           const float* __restrict__ als, const float* __restrict__ ald,
                           float* logit, float* m, int nE) {
    int i = blockIdx.x * blockDim.x + threadIdx.x;
    if (i >= nE * 8) return;
    int e = i >> 3, h = i & 7;
    float l = als[src[e] * 8 + h] + ald[dst[e] * 8 + h];
    l = l > 0.f ? l : 0.2f * l;
    logit[i] = l;
    atomicMaxF(&m[dst[e] * 8 + h], l);
}

__global__ void seg_exp(const int* __restrict__ dst, float* buf,
                        const float* __restrict__ m, float* s, int nE, int H) {
    int i = blockIdx.x * blockDim.x + threadIdx.x;
    if (i >= nE * H) return;
    int e = i / H, h = i - e * H;
    float ev = expf(buf[i] - m[dst[e] * H + h]);
    buf[i] = ev;
    atomicAdd(&s[dst[e] * H + h], ev);
}

__global__ void gat_agg(const int* __restrict__ src, const int* __restrict__ dst,
                        const float* __restrict__ ebuf, const float* __restrict__ s,
                        const float* __restrict__ H, float* out, int Co, int nE) {
    int e = blockIdx.x;
    if (e >= nE) return;
    int sn = src[e], dn = dst[e];
    const float* hp = H + (size_t)sn * 8 * Co;
    float* op = out + (size_t)dn * Co;
    for (int idx = threadIdx.x; idx < 8 * Co; idx += blockDim.x) {
        int h = idx / Co, c = idx - h * Co;
        float a = ebuf[e * 8 + h] / (s[dn * 8 + h] + 1e-16f);
        atomicAdd(&op[c], 0.125f * a * hp[idx]);
    }
}

__global__ void bias_act(float* out, const float* __restrict__ b, size_t total, int Co, int act) {
    size_t i = (size_t)blockIdx.x * blockDim.x + threadIdx.x;
    if (i >= total) return;
    int c = (int)(i % Co);
    float v = out[i] + b[c];
    if (act == 1) v = v > 0.f ? v : 0.f;
    out[i] = v;
}

__global__ void deg_kernel(const int* __restrict__ dst, float* deg, int nE) {
    int i = blockIdx.x * blockDim.x + threadIdx.x;
    if (i < nE) atomicAdd(&deg[dst[i]], 1.f);
}

__global__ void gcn_agg(const int* __restrict__ src, const int* __restrict__ dst,
                        const float* __restrict__ deg, const float* __restrict__ H,
                        float* out, int Co, int nE) {
    int e = blockIdx.x;
    if (e >= nE) return;
    int sn = src[e], dn = dst[e];
    float ds_ = deg[sn], dd = deg[dn];
    float norm = (ds_ > 0.f ? 1.f / sqrtf(ds_) : 0.f) * (dd > 0.f ? 1.f / sqrtf(dd) : 0.f);
    const float* hp = H + (size_t)sn * Co;
    float* op = out + (size_t)dn * Co;
    for (int c = threadIdx.x; c < Co; c += blockDim.x)
        atomicAdd(&op[c], norm * hp[c]);
}

__global__ void trans_logits(const int* __restrict__ ei, const float* __restrict__ q,
                             const float* __restrict__ k, float* logit, float* m, int E) {
    int gt = blockIdx.x * blockDim.x + threadIdx.x;
    int w = gt >> 5, lane = gt & 31;
    if (w >= E * 4) return;
    int e = w >> 2, h = w & 3;
    int sn = ei[e], dn = ei[E + e];
    const float* qp = q + (size_t)dn * 2048 + h * 512;
    const float* kp = k + (size_t)sn * 2048 + h * 512;
    float p = 0.f;
    for (int c = lane; c < 512; c += 32) p += qp[c] * kp[c];
    p = wave_reduce_sum(p);
    if (lane == 0) {
        float l = p * 0.044194173824159216f; // 1/sqrt(512)
        logit[w] = l;
        atomicMaxF(&m[dn * 4 + h], l);
    }
}

__global__ void trans_agg(const int* __restrict__ ei, const float* __restrict__ ebuf,
                          const float* __restrict__ s, const float* __restrict__ v,
                          float* agg, int E) {
    int e = blockIdx.x;
    if (e >= E) return;
    int sn = ei[e], dn = ei[E + e];
    const float* vp = v + (size_t)sn * 2048;
    float* op = agg + (size_t)dn * 512;
    for (int idx = threadIdx.x; idx < 2048; idx += blockDim.x) {
        int h = idx >> 9, c = idx & 511;
        float a = ebuf[e * 4 + h] / (s[dn * 4 + h] + 1e-16f);
        atomicAdd(&op[c], 0.25f * a * vp[idx]);
    }
}

__global__ void add_relu(const float* __restrict__ a, const float* __restrict__ b,
                         float* o, size_t n) {
    size_t i = (size_t)blockIdx.x * blockDim.x + threadIdx.x;
    if (i < n) { float v = a[i] + b[i]; o[i] = v > 0.f ? v : 0.f; }
}

__global__ void gap_acc(const float* __restrict__ in, const int* __restrict__ batch,
                        float* pool, int N, int C) {
    size_t i = (size_t)blockIdx.x * blockDim.x + threadIdx.x;
    if (i >= (size_t)N * C) return;
    int n = (int)(i / C), c = (int)(i % C);
    atomicAdd(&pool[(size_t)batch[n] * C + c], in[i]);
}

__global__ void gap_cnt(const int* __restrict__ batch, float* cnt, int N) {
    int i = blockIdx.x * blockDim.x + threadIdx.x;
    if (i < N) atomicAdd(&cnt[batch[i]], 1.f);
}

__global__ void gap_fin(const float* __restrict__ pool, const float* __restrict__ cnt,
                        float* out, int Bn, int C, int ldc) {
    int i = blockIdx.x * blockDim.x + threadIdx.x;
    if (i >= Bn * C) return;
    int b = i / C, c = i - b * C;
    out[(size_t)b * ldc + c] = pool[i] / fmaxf(cnt[b], 1.f);
}

__global__ void head_final(const float* __restrict__ h, const float* __restrict__ w,
                           const float* __restrict__ b, float* out, int Bn) {
    int i = blockIdx.x * blockDim.x + threadIdx.x;
    if (i >= Bn) return;
    float s = b[0];
    for (int j = 0; j < 64; j++) s += h[i * 64 + j] * w[j];
    out[i] = 1.f / (1.f + expf(-s));
}

// ---------------------------------------------------------------------------
// Host orchestration
// ---------------------------------------------------------------------------
extern "C" void kernel_launch(void* const* d_in, const int* in_sizes, int n_in,
                              void* d_out, int out_size, void* d_ws, size_t ws_size,
                              hipStream_t stream) {
    const float* x   = (const float*)d_in[0];
    const float* fp  = (const float*)d_in[1];
    const int*   ei  = (const int*)d_in[2];
    const int*   bat = (const int*)d_in[3];
    const float* g1_w = (const float*)d_in[4];  const float* g1_as = (const float*)d_in[5];
    const float* g1_ad = (const float*)d_in[6]; const float* g1_b  = (const float*)d_in[7];
    const float* g2_w = (const float*)d_in[8];  const float* g2_as = (const float*)d_in[9];
    const float* g2_ad = (const float*)d_in[10];const float* g2_b  = (const float*)d_in[11];
    const float* g3_w = (const float*)d_in[12]; const float* g3_as = (const float*)d_in[13];
    const float* g3_ad = (const float*)d_in[14];const float* g3_b  = (const float*)d_in[15];
    const float* gcn_w = (const float*)d_in[16];const float* gcn_b = (const float*)d_in[17];
    const float* t_wq = (const float*)d_in[18]; const float* t_wk = (const float*)d_in[19];
    const float* t_wv = (const float*)d_in[20];
    const float* t_bq = (const float*)d_in[21]; const float* t_bk = (const float*)d_in[22];
    const float* t_bv = (const float*)d_in[23];
    const float* t_ws = (const float*)d_in[24]; const float* t_bs = (const float*)d_in[25];
    const float* fc1_w = (const float*)d_in[26];const float* fc1_b = (const float*)d_in[27];
    const float* fc2_w = (const float*)d_in[28];const float* fc2_b = (const float*)d_in[29];
    const float* b1_w = (const float*)d_in[30]; const float* b1_b = (const float*)d_in[31];
    const float* b2_w = (const float*)d_in[32]; const float* b2_b = (const float*)d_in[33];
    const float* b3_w = (const float*)d_in[34]; const float* b3_b = (const float*)d_in[35];
    const float* b4_w = (const float*)d_in[36]; const float* b4_b = (const float*)d_in[37];

    const int N_ = in_sizes[0] / 78;     // 4096
    const int E_ = in_sizes[2] / 2;      // 8192
    const int B_ = in_sizes[1] / 1489;   // 128
    const int Esl = E_ + N_;

    // ---- workspace carve ----
    char* wsp = (char*)d_ws;
    size_t off = 0;
    auto carve = [&](size_t bytes) -> void* {
        void* p = wsp + off;
        off = (off + bytes + 255) & ~(size_t)255;
        return p;
    };
    int*   src_sl = (int*)carve((size_t)Esl * 4);
    int*   dst_sl = (int*)carve((size_t)Esl * 4);
    float* Hbuf   = (float*)carve((size_t)N_ * 16384 * 4);
    float* xa     = (float*)carve((size_t)N_ * 2048 * 4);
    float* xb     = (float*)carve((size_t)N_ * 2048 * 4);
    float* als    = (float*)carve((size_t)N_ * 8 * 4);
    float* ald    = (float*)carve((size_t)N_ * 8 * 4);
    float* mbuf   = (float*)carve((size_t)N_ * 8 * 4);
    float* sbuf   = (float*)carve((size_t)N_ * 8 * 4);
    float* alpha  = (float*)carve((size_t)Esl * 8 * 4);
    float* deg    = (float*)carve((size_t)N_ * 4);
    float* skip   = (float*)carve((size_t)N_ * 512 * 4);
    float* xc     = (float*)carve((size_t)B_ * 1024 * 4);
    float* fc1o   = (float*)carve((size_t)B_ * 128 * 4);
    float* h1     = (float*)carve((size_t)B_ * 256 * 4);
    float* h2     = (float*)carve((size_t)B_ * 128 * 4);
    float* h3     = (float*)carve((size_t)B_ * 64 * 4);
    float* pool   = (float*)carve((size_t)B_ * 512 * 4);
    float* cnt    = (float*)carve((size_t)B_ * 4);
    unsigned short* abuf = (unsigned short*)carve((size_t)N_ * 2048 * 2);     // bf16 A staging
    unsigned short* wtbuf = (unsigned short*)carve((size_t)8192 * 2048 * 2);  // bf16 W^T staging
    if (off > ws_size) return;

    auto fill = [&](float* p, float v, size_t n) {
        fillf<<<(unsigned)((n + 255) / 256), 256, 0, stream>>>(p, v, n);
    };
    // GEMM launcher: converts A (bf16, K padded to Ka) and W (bf16, transposed),
    // then runs the guard-free WMMA GEMM. All M, Nc here are multiples of 64.
    auto gemm = [&](const float* A, const float* W, float* C, const float* bias,
                    int M, int Nc, int K, int ldc, int act) {
        int Ka = (K + 31) & ~31;
        size_t atot = (size_t)M * Ka;
        conv_a<<<(unsigned)((atot + 255) / 256), 256, 0, stream>>>(A, abuf, K, Ka, atot);
        dim3 tg((Nc + 31) / 32, (Ka + 31) / 32);
        conv_wt<<<tg, dim3(32, 8), 0, stream>>>(W, wtbuf, K, Nc, Ka);
        dim3 g(Nc / 64, M / 64);
        gemm_bf16_wmma<<<g, 128, 0, stream>>>(abuf, wtbuf, C, bias, Nc, Ka, ldc, act);
    };

    build_edges<<<(Esl + 255) / 256, 256, 0, stream>>>(ei, src_sl, dst_sl, E_, N_);

    // ---- GAT layer (heads = 8) ----
    auto gat = [&](const float* xin, int Ci, int Co, const float* Wg, const float* as_,
                   const float* ad_, const float* bias, float* out) {
        gemm(xin, Wg, Hbuf, nullptr, N_, 8 * Co, Ci, 8 * Co, 0);
        int nw = N_ * 8;
        gat_head_proj<<<(nw * 32 + 255) / 256, 256, 0, stream>>>(Hbuf, as_, ad_, als, ald, N_, Co);
        fill(mbuf, -1e30f, (size_t)N_ * 8);
        fill(sbuf, 0.f, (size_t)N_ * 8);
        fill(out, 0.f, (size_t)N_ * Co);
        gat_logits<<<(Esl * 8 + 255) / 256, 256, 0, stream>>>(src_sl, dst_sl, als, ald, alpha, mbuf, Esl);
        seg_exp<<<(Esl * 8 + 255) / 256, 256, 0, stream>>>(dst_sl, alpha, mbuf, sbuf, Esl, 8);
        gat_agg<<<Esl, 256, 0, stream>>>(src_sl, dst_sl, alpha, sbuf, Hbuf, out, Co, Esl);
        bias_act<<<(unsigned)(((size_t)N_ * Co + 255) / 256), 256, 0, stream>>>(out, bias, (size_t)N_ * Co, Co, 1);
    };

    gat(x,  78,   2048, g1_w, g1_as, g1_ad, g1_b, xa);   // xa: [N, 2048]
    gat(xa, 2048, 1024, g2_w, g2_as, g2_ad, g2_b, xb);   // xb: [N, 1024]
    gat(xb, 1024, 512,  g3_w, g3_as, g3_ad, g3_b, xa);   // xa: [N, 512]

    // ---- GCN ----
    gemm(xa, gcn_w, Hbuf, nullptr, N_, 256, 512, 256, 0);
    fill(deg, 0.f, (size_t)N_);
    deg_kernel<<<(Esl + 255) / 256, 256, 0, stream>>>(dst_sl, deg, Esl);
    fill(xb, 0.f, (size_t)N_ * 256);
    gcn_agg<<<Esl, 256, 0, stream>>>(src_sl, dst_sl, deg, Hbuf, xb, 256, Esl);
    bias_act<<<(unsigned)(((size_t)N_ * 256 + 255) / 256), 256, 0, stream>>>(xb, gcn_b, (size_t)N_ * 256, 256, 0);

    // ---- global mean pool (GAT branch) -> xc[:, 0:256] ----
    auto gap = [&](const float* in, int C, float* outbase, int ldc) {
        fill(pool, 0.f, (size_t)B_ * C);
        fill(cnt, 0.f, (size_t)B_);
        gap_acc<<<(unsigned)(((size_t)N_ * C + 255) / 256), 256, 0, stream>>>(in, bat, pool, N_, C);
        gap_cnt<<<(N_ + 255) / 256, 256, 0, stream>>>(bat, cnt, N_);
        gap_fin<<<(B_ * C + 255) / 256, 256, 0, stream>>>(pool, cnt, outbase, B_, C, ldc);
    };
    gap(xb, 256, xc + 0, 1024);

    // ---- TransformerConv (heads = 4, C = 512), no self-loops ----
    float* q = Hbuf;
    float* k = Hbuf + (size_t)N_ * 2048;
    float* v = Hbuf + (size_t)N_ * 4096;
    gemm(x, t_wq, q, t_bq, N_, 2048, 78, 2048, 0);
    gemm(x, t_wk, k, t_bk, N_, 2048, 78, 2048, 0);
    gemm(x, t_wv, v, t_bv, N_, 2048, 78, 2048, 0);
    gemm(x, t_ws, skip, t_bs, N_, 512, 78, 512, 0);
    fill(mbuf, -1e30f, (size_t)N_ * 4);
    fill(sbuf, 0.f, (size_t)N_ * 4);
    fill(xa, 0.f, (size_t)N_ * 512);
    trans_logits<<<(unsigned)(((size_t)E_ * 4 * 32 + 255) / 256), 256, 0, stream>>>(ei, q, k, alpha, mbuf, E_);
    seg_exp<<<(E_ * 4 + 255) / 256, 256, 0, stream>>>(ei + E_, alpha, mbuf, sbuf, E_, 4);
    trans_agg<<<E_, 256, 0, stream>>>(ei, alpha, sbuf, v, xa, E_);
    add_relu<<<(unsigned)(((size_t)N_ * 512 + 255) / 256), 256, 0, stream>>>(xa, skip, xb, (size_t)N_ * 512);
    gap(xb, 512, xc + 256, 1024);   // xc[:, 256:768]

    // ---- fingerprint MLP -> xc[:, 768:1024] ----
    gemm(fp, fc1_w, fc1o, fc1_b, B_, 128, 1489, 128, 1);
    gemm(fc1o, fc2_w, xc + 768, fc2_b, B_, 256, 128, 1024, 1);

    // ---- head ----
    gemm(xc, b1_w, h1, b1_b, B_, 256, 1024, 256, 1);
    gemm(h1, b2_w, h2, b2_b, B_, 128, 256, 128, 1);
    gemm(h2, b3_w, h3, b3_b, B_, 64, 128, 64, 1);
    head_final<<<(B_ + 127) / 128, 128, 0, stream>>>(h3, b4_w, b4_b, (float*)d_out, B_);
}
